// Head_20040317403294
// MI455X (gfx1250) — matrix operand order repaired and checked by reading, not compiled, not run on hardware
//
#include <hip/hip_runtime.h>
#include <hip/hip_bf16.h>
#include <math.h>

typedef _Float16 v16h __attribute__((ext_vector_type(16)));
typedef float    v8f  __attribute__((ext_vector_type(8)));

static constexpr int Bq = 8;
static constexpr int T  = 2048;
static constexpr int C  = 1024;
static constexpr int D  = 128;

union FragH { v16h h; uint4 q[2]; };

// ---------------------------------------------------------------------------
// Kernel 1: W [C,D] f32  ->  Wt [3][D][C] f16 (transposed, f16)
// ---------------------------------------------------------------------------
__global__ void prep_wt(const float* __restrict__ Wq,
                        const float* __restrict__ Wk,
                        const float* __restrict__ Wv,
                        _Float16* __restrict__ wt) {
    int idx = blockIdx.x * blockDim.x + threadIdx.x;
    if (idx >= 3 * D * C) return;
    int mat = idx / (D * C);
    int rem = idx % (D * C);
    int d   = rem / C;
    int c   = rem % C;
    const float* W = (mat == 0) ? Wq : (mat == 1) ? Wk : Wv;
    wt[idx] = (_Float16)W[c * D + d];
}

// ---------------------------------------------------------------------------
// Kernel 2: projections. One wave per 16-row tile of x.
//   qf,kf: [B][T][D] f16 row-major.   vt: [B][D][T] f16 (transposed).
// ---------------------------------------------------------------------------
__global__ __launch_bounds__(32) void proj_qkv(const float* __restrict__ x,
                                               const _Float16* __restrict__ wt,
                                               _Float16* __restrict__ qf,
                                               _Float16* __restrict__ kf,
                                               _Float16* __restrict__ vt) {
    const int lane = threadIdx.x;
    const int n    = lane & 15;   // A-row / B-col / D-col index
    const int hi   = lane >> 4;   // K-half selector
    const int blk  = blockIdx.x;
    const int b    = blk / (T / 16);
    const int t0   = (blk % (T / 16)) * 16;

    const float* xrow = x + (size_t)(b * T + t0 + n) * C;

    for (int mat = 0; mat < 3; ++mat) {
        v8f acc[8] = {};
        const _Float16* wbase = wt + (size_t)mat * D * C;

        for (int ks = 0; ks < C / 32; ++ks) {
            const int k0 = ks * 32;
            // A fragment (16x32 f16) from f32 x, ISA A-layout
            const float4* ap = (const float4*)(xrow + k0);
            float4 f0 = ap[hi * 2];
            float4 f1 = ap[hi * 2 + 1];
            float4 f2 = ap[4 + hi * 2];
            float4 f3 = ap[5 + hi * 2];
            v16h a;
            a[0]=(_Float16)f0.x; a[1]=(_Float16)f0.y; a[2]=(_Float16)f0.z; a[3]=(_Float16)f0.w;
            a[4]=(_Float16)f1.x; a[5]=(_Float16)f1.y; a[6]=(_Float16)f1.z; a[7]=(_Float16)f1.w;
            a[8]=(_Float16)f2.x; a[9]=(_Float16)f2.y; a[10]=(_Float16)f2.z; a[11]=(_Float16)f2.w;
            a[12]=(_Float16)f3.x; a[13]=(_Float16)f3.y; a[14]=(_Float16)f3.z; a[15]=(_Float16)f3.w;

            for (int nt = 0; nt < 8; ++nt) {
                const _Float16* bp = wbase + (size_t)(nt * 16 + n) * C + k0 + hi * 8;
                FragH ub;
                ub.q[0] = *(const uint4*)(bp);
                ub.q[1] = *(const uint4*)(bp + 16);
                acc[nt] = __builtin_amdgcn_wmma_f32_16x16x32_f16(
                    false, a, false, ub.h, (short)0, acc[nt], false, false);
            }
        }

        if (mat < 2) {  // Q / K: row-major f16 stores
            _Float16* dst = (mat == 0) ? qf : kf;
            for (int nt = 0; nt < 8; ++nt)
                for (int r = 0; r < 8; ++r)
                    dst[(size_t)(b * T + t0 + r + 8 * hi) * D + nt * 16 + n] =
                        (_Float16)acc[nt][r];
        } else {        // V: transposed store, 8 halves pack to one b128
            for (int nt = 0; nt < 8; ++nt) {
                union { uint4 q; _Float16 h[8]; } uo;
                for (int r = 0; r < 8; ++r) uo.h[r] = (_Float16)acc[nt][r];
                *(uint4*)(vt + ((size_t)(b * D) + nt * 16 + n) * T + t0 + 8 * hi) = uo.q;
            }
        }
    }
}

// ---------------------------------------------------------------------------
// Async staging helper: one b128 of K-tile + one b128 of V-tile per thread.
//   K tile: 32 keys x 128 d (f16, row-major by key)   = 8 KB
//   V tile: 128 d  x 32 keys (f16, row-major by d)    = 8 KB
// ---------------------------------------------------------------------------
__device__ inline void issue_tile_async(const _Float16* __restrict__ ksrc,
                                        const _Float16* __restrict__ vsrc,
                                        unsigned kbufOff, unsigned vbufOff,
                                        int tid) {
    {   // K: thread -> (row = tid/8, 16-half chunk = tid%8)
        const int row = tid >> 3, chunk = tid & 7;
        unsigned long long ga =
            (unsigned long long)(const void*)(ksrc + (size_t)row * D + chunk * 16);
        unsigned la = kbufOff + (unsigned)((row * 128 + chunk * 16) * 2);
        asm volatile("global_load_async_to_lds_b128 %0, %1, off"
                     :: "v"(la), "v"(ga) : "memory");
    }
    {   // V: thread -> (row = tid/2, 16-half chunk = tid%2); vt row stride = T
        const int row = tid >> 1, chunk = tid & 1;
        unsigned long long ga =
            (unsigned long long)(const void*)(vsrc + (size_t)row * T + chunk * 16);
        unsigned la = vbufOff + (unsigned)((row * 32 + chunk * 16) * 2);
        asm volatile("global_load_async_to_lds_b128 %0, %1, off"
                     :: "v"(la), "v"(ga) : "memory");
    }
}

// ---------------------------------------------------------------------------
// Kernel 3: causal flash attention.
//   8 waves / block; block owns 128 query rows; K/V tiles double-buffered in
//   LDS via async-to-LDS DMA and shared by all waves.
// ---------------------------------------------------------------------------
__global__ __launch_bounds__(256) void attn(const _Float16* __restrict__ qf,
                                            const _Float16* __restrict__ kf,
                                            const _Float16* __restrict__ vt,
                                            float* __restrict__ out) {
    __shared__ __align__(16) _Float16 kbuf[2][32 * 128];
    __shared__ __align__(16) _Float16 vbuf[2][128 * 32];
    __shared__ __align__(16) _Float16 pbuf[8][16 * 32];

    const int tid  = threadIdx.x;
    const int w    = tid >> 5;        // wave id (0..7)
    const int lane = tid & 31;
    const int n    = lane & 15;
    const int hi   = lane >> 4;
    const int blk  = blockIdx.x;
    const int b    = blk / (T / 128);
    const int t0   = (blk % (T / 128)) * 128;   // block's first query row
    const int tq   = t0 + 16 * w;               // this wave's query tile

    const unsigned kOff0 = (unsigned)(uintptr_t)&kbuf[0][0];
    const unsigned kOff1 = (unsigned)(uintptr_t)&kbuf[1][0];
    const unsigned vOff0 = (unsigned)(uintptr_t)&vbuf[0][0];
    const unsigned vOff1 = (unsigned)(uintptr_t)&vbuf[1][0];

    // Preload Q fragments (4 K-steps over D=128)
    v16h qfr[4];
    const _Float16* qrow = qf + (size_t)(b * T + tq + n) * D;
    for (int s = 0; s < 4; ++s) {
        FragH u;
        u.q[0] = *(const uint4*)(qrow + 32 * s + hi * 8);
        u.q[1] = *(const uint4*)(qrow + 32 * s + 16 + hi * 8);
        qfr[s] = u.h;
    }

    v8f acc[8] = {};
    float mrow[8], lrow[8];
    for (int r = 0; r < 8; ++r) { mrow[r] = -__builtin_inff(); lrow[r] = 0.f; }

    const float scale = 0.088388347648318447f;  // 1/sqrt(128)

    const _Float16* kbatch = kf + (size_t)b * T * D;
    const _Float16* vbatch = vt + (size_t)b * D * T;

    const int nsteps = (t0 + 128) / 32;         // keys 0 .. t0+127

    // prime the pipeline: tile 0 -> buffer 0
    issue_tile_async(kbatch, vbatch, kOff0, vOff0, tid);

    for (int step = 0; step < nsteps; ++step) {
        const int kv0 = step * 32;
        const int p   = step & 1;

        asm volatile("s_wait_asynccnt 0x0" ::: "memory");
        __syncthreads();

        if (step + 1 < nsteps)
            issue_tile_async(kbatch + (size_t)(kv0 + 32) * D,
                             vbatch + (kv0 + 32),
                             p ? kOff0 : kOff1, p ? vOff0 : vOff1, tid);

        // ---- S = Q K^T : two 16x16 tiles from the LDS K tile ----
        v8f s0 = {}, s1 = {};
        for (int j = 0; j < 2; ++j) {
            const _Float16* krow = &kbuf[p][(16 * j + n) * 128];
            for (int s = 0; s < 4; ++s) {
                FragH u;
                u.q[0] = *(const uint4*)(krow + 32 * s + hi * 8);
                u.q[1] = *(const uint4*)(krow + 32 * s + 16 + hi * 8);
                if (j == 0)
                    s0 = __builtin_amdgcn_wmma_f32_16x16x32_f16(
                        false, qfr[s], false, u.h, (short)0, s0, false, false);
                else
                    s1 = __builtin_amdgcn_wmma_f32_16x16x32_f16(
                        false, qfr[s], false, u.h, (short)0, s1, false, false);
            }
        }

        // ---- online softmax over the 32-key block ----
        float corr[8];
        for (int r = 0; r < 8; ++r) {
            const int qi = tq + r + 8 * hi;
            float a0 = s0[r] * scale; if (kv0 + n      > qi) a0 = -__builtin_inff();
            float a1 = s1[r] * scale; if (kv0 + 16 + n > qi) a1 = -__builtin_inff();
            float rm = fmaxf(a0, a1);
            rm = fmaxf(rm, __shfl_xor(rm, 1));
            rm = fmaxf(rm, __shfl_xor(rm, 2));
            rm = fmaxf(rm, __shfl_xor(rm, 4));
            rm = fmaxf(rm, __shfl_xor(rm, 8));
            const float mnew = fmaxf(mrow[r], rm);
            const float c    = __expf(mrow[r] - mnew);
            const float p0   = __expf(a0 - mnew);
            const float p1   = __expf(a1 - mnew);
            float rs = p0 + p1;
            rs += __shfl_xor(rs, 1);
            rs += __shfl_xor(rs, 2);
            rs += __shfl_xor(rs, 4);
            rs += __shfl_xor(rs, 8);
            lrow[r] = lrow[r] * c + rs;
            mrow[r] = mnew;
            corr[r] = c;
            // stage P into this wave's LDS tile (D-layout -> row-major 16x32)
            pbuf[w][(r + 8 * hi) * 32 + n]      = (_Float16)p0;
            pbuf[w][(r + 8 * hi) * 32 + 16 + n] = (_Float16)p1;
        }
        for (int dt = 0; dt < 8; ++dt)
            for (int r = 0; r < 8; ++r) acc[dt][r] *= corr[r];

        asm volatile("s_wait_dscnt 0x0" ::: "memory");

        // reload P as A-fragment (16x32 f16)
        FragH up;
        up.q[0] = *(const uint4*)(&pbuf[w][n * 32 + hi * 8]);
        up.q[1] = *(const uint4*)(&pbuf[w][n * 32 + 16 + hi * 8]);

        // ---- O += P V : 8 d-tiles from the LDS V tile ----
        for (int dt = 0; dt < 8; ++dt) {
            const _Float16* vrow = &vbuf[p][(dt * 16 + n) * 32 + hi * 8];
            FragH uv;
            uv.q[0] = *(const uint4*)(vrow);
            uv.q[1] = *(const uint4*)(vrow + 16);
            acc[dt] = __builtin_amdgcn_wmma_f32_16x16x32_f16(
                false, up.h, false, uv.h, (short)0, acc[dt], false, false);
        }
    }

    // ---- epilogue: normalize and store f32 output ----
    for (int dt = 0; dt < 8; ++dt)
        for (int r = 0; r < 8; ++r)
            out[(size_t)(b * T + tq + r + 8 * hi) * D + dt * 16 + n] =
                acc[dt][r] / lrow[r];
}

// ---------------------------------------------------------------------------
extern "C" void kernel_launch(void* const* d_in, const int* in_sizes, int n_in,
                              void* d_out, int out_size, void* d_ws, size_t ws_size,
                              hipStream_t stream) {
    const float* x  = (const float*)d_in[0];
    const float* Wq = (const float*)d_in[1];
    const float* Wk = (const float*)d_in[2];
    const float* Wv = (const float*)d_in[3];
    float* out = (float*)d_out;

    _Float16* wt = (_Float16*)d_ws;                       // 3*D*C halves
    _Float16* qf = wt + (size_t)3 * D * C;                // B*T*D halves
    _Float16* kf = qf + (size_t)Bq * T * D;
    _Float16* vt = kf + (size_t)Bq * T * D;

    prep_wt<<<(3 * D * C + 255) / 256, 256, 0, stream>>>(Wq, Wk, Wv, wt);
    proj_qkv<<<Bq * T / 16, 32, 0, stream>>>(x, wt, qf, kf, vt);
    attn<<<Bq * T / 128, 256, 0, stream>>>(qf, kf, vt, out);
}
